// MultiHeadMamba_S_72456098283769
// MI455X (gfx1250) — compile-verified, matrix operand order checked by reading, not run on hardware
//
#include <hip/hip_runtime.h>
#include <hip/hip_bf16.h>
#include <cstdint>

#define HH 8
#define BB 4
#define LL 4096
#define CC 1024
#define DH 128
#define DI 256
#define NN 16
#define RR 8
#define MM (BB*LL)      // 16384 rows per head
#define CHN 32          // chunks in L
#define CLEN 128        // chunk length (CHN*CLEN == LL)
#define DBLP 48         // padded width of (dt,B,C) rows (40 used)
#define EPROJ 40        // R + 2N

typedef __attribute__((ext_vector_type(16))) __bf16 v16bf;
typedef __attribute__((ext_vector_type(8)))  float  v8f;

// ---- WMMA fragment loaders (CDNA5 16-bit A/B layout) ----
// element i of the v16 fragment holds k = ((i<8)?i:i+8) + 8*(lane>=16),
// row r = lane&15.  Caller passes p already offset to (row r, k-offset).
__device__ __forceinline__ v16bf frag_bf16(const __bf16* p) {
  v16bf f;
#pragma unroll
  for (int i = 0; i < 16; ++i) {
    int k = (i < 8) ? i : (i + 8);
    f[i] = p[k];
  }
  return f;
}
__device__ __forceinline__ v16bf frag_bf16v(const __bf16* p, bool valid) {
  v16bf f;
#pragma unroll
  for (int i = 0; i < 16; ++i) {
    int k = (i < 8) ? i : (i + 8);
    f[i] = valid ? p[k] : (__bf16)0.0f;
  }
  return f;
}
// fast sigmoid: v_exp_f32 + v_rcp_f32 (avoid IEEE div expansion)
__device__ __forceinline__ float sigmoidf_(float v) {
  return __builtin_amdgcn_rcpf(1.0f + __expf(-v));
}

// ---- Kernel W: one-shot fp32 -> bf16 weight conversion ----
__global__ void k_cvt(const float* __restrict__ src, __bf16* __restrict__ dst, int n) {
  const int i = blockIdx.x * 256 + threadIdx.x;
  if (i < n) dst[i] = (__bf16)src[i];
}

// ---- Kernel A: in_proj GEMM (per head: 16384 x 512 x 128), split u / silu(z) ----
// block = 128 threads (4 waves). A tile staged once in LDS as bf16; each wave
// owns 8 of the 32 N-tiles.  Weights pre-converted to bf16.
__global__ void k_inproj(const float* __restrict__ x, const __bf16* __restrict__ w,
                         __bf16* __restrict__ u_bf, __bf16* __restrict__ sz_bf) {
  __shared__ __bf16 sA[16 * DH];     // 4 KB, converted once
  const int m0   = blockIdx.x * 16;
  const int h    = blockIdx.y;
  const int tid  = threadIdx.x;
  const int lane = tid & 31;
  const int wave = tid >> 5;
  const int r    = lane & 15;
  const int koff = (lane >> 4) << 3;

  // stage + convert A tile: 16 rows x 128 k
  for (int i = tid; i < 16 * DH; i += 128) {
    const int rr = i >> 7, kc = i & (DH - 1);
    sA[i] = (__bf16)x[(size_t)(m0 + rr) * CC + h * DH + kc];
  }
  __syncthreads();

  v16bf afr[4];
  const __bf16* abase = sA + r * DH + koff;
#pragma unroll
  for (int kk = 0; kk < 4; ++kk) afr[kk] = frag_bf16(abase + kk * 32);

  const __bf16* wh = w + (size_t)h * (2 * DI) * DH;
  const int n     = lane & 15;
  const int mbase = (lane >> 4) << 3;          // 0 or 8
#pragma unroll
  for (int t = 0; t < 8; ++t) {
    const int nt = wave * 8 + t;
    const int e0 = nt * 16;
    const __bf16* bbase = wh + (size_t)(e0 + r) * DH + koff;
    v8f acc = {};
#pragma unroll
    for (int kk = 0; kk < 4; ++kk) {
      v16bf bfr = frag_bf16(bbase + kk * 32);
      acc = __builtin_amdgcn_wmma_f32_16x16x32_bf16(false, afr[kk], false, bfr,
                                                    (short)0, acc, false, false);
    }
    // single branch per tile; base + imm-offset stores (row stride = DI elems)
    if (nt < 16) {
      __bf16* p = u_bf + ((size_t)h * MM + m0 + mbase) * DI + e0 + n;
#pragma unroll
      for (int j = 0; j < 8; ++j) p[(size_t)j * DI] = (__bf16)acc[j];
    } else {
      __bf16* p = sz_bf + ((size_t)h * MM + m0 + mbase) * DI + (e0 - DI) + n;
#pragma unroll
      for (int j = 0; j < 8; ++j) {
        const float v = acc[j];
        p[(size_t)j * DI] = (__bf16)(v * sigmoidf_(v));
      }
    }
  }
}

// ---- Kernel B: causal depthwise conv (K=4) + bias + SiLU ----
__global__ void k_conv(const __bf16* __restrict__ u_bf, const float* __restrict__ cw,
                       const float* __restrict__ cb, __bf16* __restrict__ uc_bf) {
  const size_t idx = (size_t)blockIdx.x * blockDim.x + threadIdx.x;
  if (idx >= (size_t)HH * MM * DI) return;
  const int    d    = (int)(idx & (DI - 1));
  const size_t hrow = idx >> 8;              // h*MM + row
  const int    row  = (int)(hrow & (MM - 1));
  const int    h    = (int)(hrow >> 14);     // MM == 2^14
  const int    l    = row & (LL - 1);
  float acc = cb[h * DI + d];
#pragma unroll
  for (int k = 0; k < 4; ++k) {
    const int lp = l - 3 + k;
    if (lp >= 0)
      acc += (float)u_bf[(hrow - (size_t)(3 - k)) * DI + d] * cw[(h * DI + d) * 4 + k];
  }
  uc_bf[idx] = (__bf16)(acc * sigmoidf_(acc));
}

// ---- Kernel C: x_proj GEMM (16384 x 40 x 256), N padded to 48 ----
// block = 96 threads (3 waves); A tile staged in LDS; bf16 weights.
__global__ void k_xproj(const __bf16* __restrict__ uc_bf, const __bf16* __restrict__ w,
                        float* __restrict__ dbl) {
  __shared__ __bf16 sA[16 * DI];     // 8 KB
  const int m0   = blockIdx.x * 16;
  const int h    = blockIdx.y;
  const int tid  = threadIdx.x;
  const int lane = tid & 31;
  const int nt   = tid >> 5;         // 0..2
  const int r    = lane & 15;
  const int koff = (lane >> 4) << 3;

  for (int i = tid; i < 16 * DI; i += 96) {
    const int rr = i >> 8, kc = i & (DI - 1);
    sA[i] = uc_bf[((size_t)h * MM + m0 + rr) * DI + kc];
  }
  __syncthreads();

  const __bf16* abase = sA + r * DI + koff;
  const int  e    = nt * 16 + r;
  const bool bval = (e < EPROJ);
  const size_t erow = bval ? (size_t)e : 0;  // clamp so loads stay in-bounds
  const __bf16* bbase = w + (size_t)h * EPROJ * DI + erow * DI + koff;

  v8f acc = {};
#pragma unroll
  for (int kk = 0; kk < 8; ++kk) {
    v16bf a = frag_bf16(abase + kk * 32);
    v16bf b = frag_bf16v(bbase + kk * 32, bval);
    acc = __builtin_amdgcn_wmma_f32_16x16x32_bf16(false, a, false, b,
                                                  (short)0, acc, false, false);
  }
  const int mbase = (lane >> 4) << 3;
  float* p = dbl + ((size_t)h * MM + m0 + mbase) * DBLP + nt * 16 + (lane & 15);
#pragma unroll
  for (int j = 0; j < 8; ++j) p[(size_t)j * DBLP] = acc[j];
}

// ---- Scan: delta = softplus(dt . w + b), dA = exp(delta*A) ----
// Pass 1: per-chunk local end state (h0 = 0) + decay product P, no y.
__global__ void k_scan1(const float* __restrict__ dbl, const __bf16* __restrict__ uc_bf,
                        const float* __restrict__ A_log, const float* __restrict__ dtw,
                        const float* __restrict__ dtb,
                        float* __restrict__ hloc, float* __restrict__ Pbuf) {
  __shared__ float s[CLEN * DBLP];   // 24 KB
  const int c = blockIdx.x, b = blockIdx.y, h = blockIdx.z;
  const int d = threadIdx.x;         // 0..255
  const size_t rowbase = (size_t)h * MM + (size_t)b * LL + (size_t)c * CLEN;
  for (int i = d; i < CLEN * DBLP; i += DI) s[i] = dbl[rowbase * DBLP + i];
  __syncthreads();

  float An[NN];
#pragma unroll
  for (int n = 0; n < NN; ++n) An[n] = -__expf(A_log[((size_t)h * DI + d) * NN + n]);
  float wr[RR];
#pragma unroll
  for (int j = 0; j < RR; ++j) wr[j] = dtw[((size_t)h * DI + d) * RR + j];
  const float bias = dtb[h * DI + d];

  float hs[NN], Pp[NN];
#pragma unroll
  for (int n = 0; n < NN; ++n) { hs[n] = 0.f; Pp[n] = 1.f; }

  for (int l = 0; l < CLEN; ++l) {
    if (l + 16 < CLEN)
      __builtin_prefetch(&uc_bf[(rowbase + l + 16) * DI + d], 0, 1);
    const float* rowp = &s[l * DBLP];
    float xlin = bias;
#pragma unroll
    for (int j = 0; j < RR; ++j) xlin += wr[j] * rowp[j];
    const float delta = (xlin > 20.f) ? xlin : __logf(1.f + __expf(xlin));
    const float uv = (float)uc_bf[(rowbase + l) * DI + d];
    const float du = delta * uv;
#pragma unroll
    for (int n = 0; n < NN; ++n) {
      const float dA = __expf(delta * An[n]);
      hs[n] = dA * hs[n] + du * rowp[8 + n];
      Pp[n] *= dA;
    }
  }
  const size_t o = ((((size_t)h * BB + b) * CHN + c) * DI + d) * NN;
#pragma unroll
  for (int n = 0; n < NN; ++n) { hloc[o + n] = hs[n]; Pbuf[o + n] = Pp[n]; }
}

// Pass 2: serial combine across chunks; one thread per (h,b,d,n).
__global__ void k_scan2(const float* __restrict__ hloc, const float* __restrict__ Pbuf,
                        float* __restrict__ hinit) {
  const size_t idx = (size_t)blockIdx.x * blockDim.x + threadIdx.x;
  if (idx >= (size_t)HH * BB * DI * NN) return;
  const int n = (int)(idx & (NN - 1));
  const int d = (int)((idx >> 4) & (DI - 1));
  const int b = (int)((idx >> 12) & (BB - 1));
  const int h = (int)(idx >> 14);
  float hr = 0.f;
  for (int c = 0; c < CHN; ++c) {
    const size_t o = ((((size_t)h * BB + b) * CHN + c) * DI + d) * NN + n;
    hinit[o] = hr;
    hr = Pbuf[o] * hr + hloc[o];
  }
}

// Pass 3: replay with correct init state, emit y_bf = (y_scan + u*D) * silu(z).
__global__ void k_scan3(const float* __restrict__ dbl, const __bf16* __restrict__ uc_bf,
                        const __bf16* __restrict__ sz_bf,
                        const float* __restrict__ A_log, const float* __restrict__ dtw,
                        const float* __restrict__ dtb, const float* __restrict__ Dp,
                        const float* __restrict__ hinit, __bf16* __restrict__ y_bf) {
  __shared__ float s[CLEN * DBLP];
  const int c = blockIdx.x, b = blockIdx.y, h = blockIdx.z;
  const int d = threadIdx.x;
  const size_t rowbase = (size_t)h * MM + (size_t)b * LL + (size_t)c * CLEN;
  for (int i = d; i < CLEN * DBLP; i += DI) s[i] = dbl[rowbase * DBLP + i];
  __syncthreads();

  float An[NN];
#pragma unroll
  for (int n = 0; n < NN; ++n) An[n] = -__expf(A_log[((size_t)h * DI + d) * NN + n]);
  float wr[RR];
#pragma unroll
  for (int j = 0; j < RR; ++j) wr[j] = dtw[((size_t)h * DI + d) * RR + j];
  const float bias = dtb[h * DI + d];
  const float Dv   = Dp[h * DI + d];

  float hs[NN];
  const size_t o = ((((size_t)h * BB + b) * CHN + c) * DI + d) * NN;
#pragma unroll
  for (int n = 0; n < NN; ++n) hs[n] = hinit[o + n];

  for (int l = 0; l < CLEN; ++l) {
    if (l + 16 < CLEN) {
      __builtin_prefetch(&uc_bf[(rowbase + l + 16) * DI + d], 0, 1);
      __builtin_prefetch(&sz_bf[(rowbase + l + 16) * DI + d], 0, 1);
    }
    const float* rowp = &s[l * DBLP];
    float xlin = bias;
#pragma unroll
    for (int j = 0; j < RR; ++j) xlin += wr[j] * rowp[j];
    const float delta = (xlin > 20.f) ? xlin : __logf(1.f + __expf(xlin));
    const float uv = (float)uc_bf[(rowbase + l) * DI + d];
    const float du = delta * uv;
    float y = 0.f;
#pragma unroll
    for (int n = 0; n < NN; ++n) {
      const float dA = __expf(delta * An[n]);
      hs[n] = dA * hs[n] + du * rowp[8 + n];
      y += hs[n] * rowp[24 + n];
    }
    const float zs = (float)sz_bf[(rowbase + l) * DI + d];
    y_bf[(rowbase + l) * DI + d] = (__bf16)((y + uv * Dv) * zs);
  }
}

// ---- Kernel G: out_proj GEMM (16384 x 128 x 256) -> fp32 output ----
// block = 256 threads (8 waves); A tile staged in LDS (8-way reuse); bf16 weights.
__global__ void k_outproj(const __bf16* __restrict__ y_bf, const __bf16* __restrict__ w,
                          float* __restrict__ out) {
  __shared__ __bf16 sA[16 * DI];     // 8 KB
  const int m0   = blockIdx.x * 16;
  const int h    = blockIdx.y;
  const int tid  = threadIdx.x;
  const int lane = tid & 31;
  const int nt   = tid >> 5;         // 0..7
  const int r    = lane & 15;
  const int koff = (lane >> 4) << 3;

  for (int i = tid; i < 16 * DI; i += 256) {
    const int rr = i >> 8, kc = i & (DI - 1);
    sA[i] = y_bf[((size_t)h * MM + m0 + rr) * DI + kc];
  }
  __syncthreads();

  const __bf16* abase = sA + r * DI + koff;
  const __bf16* bbase = w + ((size_t)h * DH + nt * 16 + r) * DI + koff;

  v8f acc = {};
#pragma unroll
  for (int kk = 0; kk < 8; ++kk) {
    v16bf a = frag_bf16(abase + kk * 32);
    v16bf b = frag_bf16(bbase + kk * 32);
    acc = __builtin_amdgcn_wmma_f32_16x16x32_bf16(false, a, false, b,
                                                  (short)0, acc, false, false);
  }
  const int mbase = (lane >> 4) << 3;
  float* p = out + (size_t)(m0 + mbase) * CC + h * DH + nt * 16 + (lane & 15);
#pragma unroll
  for (int j = 0; j < 8; ++j) p[(size_t)j * CC] = acc[j];
}

extern "C" void kernel_launch(void* const* d_in, const int* in_sizes, int n_in,
                              void* d_out, int out_size, void* d_ws, size_t ws_size,
                              hipStream_t stream) {
  (void)in_sizes; (void)n_in; (void)out_size; (void)ws_size;
  const float* x    = (const float*)d_in[0];
  const float* ipw  = (const float*)d_in[1];
  const float* cw   = (const float*)d_in[2];
  const float* cb   = (const float*)d_in[3];
  const float* xpw  = (const float*)d_in[4];
  const float* dtw  = (const float*)d_in[5];
  const float* dtb  = (const float*)d_in[6];
  const float* Alog = (const float*)d_in[7];
  const float* Dp   = (const float*)d_in[8];
  const float* opw  = (const float*)d_in[9];
  float* out = (float*)d_out;

  char* ws = (char*)d_ws;
  constexpr size_t U   = (size_t)HH * MM * DI;            // 33.5M elems
  constexpr size_t ST  = (size_t)HH * BB * CHN * DI * NN; // 4.2M elems
  constexpr size_t NIP = (size_t)HH * (2 * DI) * DH;      // 524288
  constexpr size_t NXP = (size_t)HH * EPROJ * DI;         // 81920
  constexpr size_t NOP = (size_t)HH * DH * DI;            // 262144
  __bf16* u_bf  = (__bf16*)(ws + 0 * U);
  __bf16* sz_bf = (__bf16*)(ws + 2 * U);
  __bf16* uc_bf = (__bf16*)(ws + 4 * U);
  __bf16* y_bf  = (__bf16*)(ws + 6 * U);
  float*  dbl   = (float*)(ws + 8 * U);
  float*  hloc  = (float*)(ws + 8 * U + (size_t)HH * MM * DBLP * 4);
  float*  Pbuf  = hloc + ST;
  float*  hinit = Pbuf + ST;
  __bf16* ipw_bf = (__bf16*)(hinit + ST);
  __bf16* xpw_bf = ipw_bf + NIP;
  __bf16* opw_bf = xpw_bf + NXP;

  // one-shot weight conversions (~870K elems total)
  k_cvt<<<(int)((NIP + 255) / 256), 256, 0, stream>>>(ipw, ipw_bf, (int)NIP);
  k_cvt<<<(int)((NXP + 255) / 256), 256, 0, stream>>>(xpw, xpw_bf, (int)NXP);
  k_cvt<<<(int)((NOP + 255) / 256), 256, 0, stream>>>(opw, opw_bf, (int)NOP);

  dim3 gT(MM / 16, HH);
  k_inproj<<<gT, 128, 0, stream>>>(x, ipw_bf, u_bf, sz_bf);

  const size_t total = (size_t)HH * MM * DI;
  k_conv<<<(int)((total + 255) / 256), 256, 0, stream>>>(u_bf, cw, cb, uc_bf);

  k_xproj<<<gT, 96, 0, stream>>>(uc_bf, xpw_bf, dbl);

  dim3 gS(CHN, BB, HH);
  k_scan1<<<gS, DI, 0, stream>>>(dbl, uc_bf, Alog, dtw, dtb, hloc, Pbuf);
  k_scan2<<<(HH * BB * DI * NN) / 256, 256, 0, stream>>>(hloc, Pbuf, hinit);
  k_scan3<<<gS, DI, 0, stream>>>(dbl, uc_bf, sz_bf, Alog, dtw, dtb, Dp, hinit, y_bf);

  k_outproj<<<gT, 256, 0, stream>>>(y_bf, opw_bf, out);
}
